// RNN_26843545600543
// MI455X (gfx1250) — compile-verified, matrix operand order checked by reading, not run on hardware
//
#include <hip/hip_runtime.h>
#include <hip/hip_bf16.h>

// MI455X / gfx1250 persistent 3-layer LSTM.
// Grid: 32 blocks x 256 threads (8 waves). Each block owns a 16-row batch tile
// and runs all T=2048 steps x 3 layers with weight WMMA B-fragments resident in
// VGPRs, h-state (f16) in LDS, c-state in thread-private registers.

typedef __attribute__((ext_vector_type(16))) _Float16 v16h;
typedef __attribute__((ext_vector_type(8)))  float    v8f;

#define HDIM   64
#define TSEQ   2048
#define BATCH  512
#define NOUT   11
#define TILE_B 16
#define NGATE  256   // 4*H

__device__ __forceinline__ float fast_sigmoid(float x) {
    // 1 / (1 + 2^(-x*log2(e)))  -> v_exp_f32 + v_rcp_f32
    float e = __builtin_amdgcn_exp2f(-x * 1.442695040888963f);
    return __builtin_amdgcn_rcpf(1.0f + e);
}
__device__ __forceinline__ float fast_tanh(float x) {
    // 1 - 2/(2^(2x*log2(e)) + 1)
    float e = __builtin_amdgcn_exp2f(x * 2.885390081777927f);
    return 1.0f - 2.0f * __builtin_amdgcn_rcpf(e + 1.0f);
}

// Build a WMMA B fragment (K=32 chunk) from a row-major [256][64] fp32 weight
// matrix, transposed+converted on the fly. Lane layout mirrors the 16-bit
// 16x32 A layout: lane = col (0..15), lane>>4 selects the K half-stripe.
__device__ __forceinline__ v16h load_w_frag(const float* __restrict__ W,
                                            int tile, int kc, int lane) {
    int col = tile * 16 + (lane & 15);
    int kb  = kc * 32 + (lane >> 4) * 8;
    const float* p = W + col * HDIM + kb;
    v16h f;
#pragma unroll
    for (int i = 0; i < 8; ++i) f[i] = (_Float16)p[i];
#pragma unroll
    for (int i = 0; i < 8; ++i) f[8 + i] = (_Float16)p[16 + i];
    return f;
}

// Build A fragment (16x32 f16, ISA layout) from LDS h-state [16][64] row-major.
__device__ __forceinline__ v16h load_a_frag(const _Float16* __restrict__ hlds,
                                            int kc, int lane) {
    int row = lane & 15;
    int kb  = kc * 32 + (lane >> 4) * 8;
    const _Float16* p = hlds + row * HDIM + kb;
    v16h f;
#pragma unroll
    for (int i = 0; i < 8; ++i) f[i] = p[i];
#pragma unroll
    for (int i = 0; i < 8; ++i) f[8 + i] = p[16 + i];
    return f;
}

__global__ __launch_bounds__(256, 1)
void lstm3_wmma_persistent(
    const float* __restrict__ x,
    const float* __restrict__ Wih0, const float* __restrict__ Whh0,
    const float* __restrict__ bih0, const float* __restrict__ bhh0,
    const float* __restrict__ Wih1, const float* __restrict__ Whh1,
    const float* __restrict__ bih1, const float* __restrict__ bhh1,
    const float* __restrict__ Wih2, const float* __restrict__ Whh2,
    const float* __restrict__ bih2, const float* __restrict__ bhh2,
    const float* __restrict__ Wout, const float* __restrict__ bout,
    float* __restrict__ out)
{
    __shared__ float    gates[NGATE * TILE_B];      // [col][row], 16 KB
    __shared__ _Float16 hst[3][TILE_B * HDIM];      // per-layer h (t-1 / t), 6 KB
    __shared__ float    biasc[3][NGATE];            // bih+bhh, 3 KB
    __shared__ float    wi0[NGATE * 2];             // Wih0, 2 KB
    __shared__ float    xst[TILE_B * 2];            // staged x_t

    const int tid  = threadIdx.x;
    const int lane = tid & 31;
    const int wave = tid >> 5;
    const int b0   = blockIdx.x * TILE_B;

    // ---- stage biases, Wih0; zero h-state ----
    for (int i = tid; i < NGATE; i += 256) {
        biasc[0][i] = bih0[i] + bhh0[i];
        biasc[1][i] = bih1[i] + bhh1[i];
        biasc[2][i] = bih2[i] + bhh2[i];
        wi0[i * 2 + 0] = Wih0[i * 2 + 0];
        wi0[i * 2 + 1] = Wih0[i * 2 + 1];
    }
    for (int i = tid; i < TILE_B * HDIM; i += 256) {
        hst[0][i] = (_Float16)0.f;
        hst[1][i] = (_Float16)0.f;
        hst[2][i] = (_Float16)0.f;
    }

    // ---- weight fragments resident in VGPRs for the whole sequence ----
    v16h whhF[3][2][2];   // [layer][tile][kchunk]
    v16h wihF[2][2][2];   // layers 1,2 input weights (fused as extra K chunks)
    {
        const float* Whh[3] = { Whh0, Whh1, Whh2 };
        const float* Wih[2] = { Wih1, Wih2 };
#pragma unroll
        for (int l = 0; l < 3; ++l)
#pragma unroll
            for (int ti = 0; ti < 2; ++ti)
#pragma unroll
                for (int kc = 0; kc < 2; ++kc)
                    whhF[l][ti][kc] = load_w_frag(Whh[l], wave * 2 + ti, kc, lane);
#pragma unroll
        for (int l = 0; l < 2; ++l)
#pragma unroll
            for (int ti = 0; ti < 2; ++ti)
#pragma unroll
                for (int kc = 0; kc < 2; ++kc)
                    wihF[l][ti][kc] = load_w_frag(Wih[l], wave * 2 + ti, kc, lane);
    }

    // thread-private cell state: this thread owns 4 consecutive (row,j) pairs
    float cpr[3][4];
#pragma unroll
    for (int l = 0; l < 3; ++l)
#pragma unroll
        for (int i = 0; i < 4; ++i) cpr[l][i] = 0.f;

    const int idx0 = tid * 4;         // 1024 (row,j) pairs / 256 threads
    const int crow = idx0 >> 6;       // 0..15
    const int cj   = idx0 & 63;       // 0..60 step 4

    __syncthreads();

    for (int t = 0; t < TSEQ; ++t) {
        // stage x_t for this batch tile
        if (tid < TILE_B * 2) {
            int row = tid >> 1, ch = tid & 1;
            xst[tid] = x[((size_t)(b0 + row) * TSEQ + t) * 2 + ch];
        }

#pragma unroll
        for (int l = 0; l < 3; ++l) {
            // ---- gate matmul: [16,64(+64)] x [64(+64),256] via WMMA ----
#pragma unroll
            for (int ti = 0; ti < 2; ++ti) {
                v8f acc = {};
#pragma unroll
                for (int kc = 0; kc < 2; ++kc) {
                    v16h a = load_a_frag(hst[l], kc, lane);       // h_l(t-1)
                    acc = __builtin_amdgcn_wmma_f32_16x16x32_f16(
                        false, a, false, whhF[l][ti][kc], (short)0, acc,
                        false, false);
                }
                if (l > 0) {
#pragma unroll
                    for (int kc = 0; kc < 2; ++kc) {
                        v16h a = load_a_frag(hst[l - 1], kc, lane); // h_{l-1}(t)
                        acc = __builtin_amdgcn_wmma_f32_16x16x32_f16(
                            false, a, false, wihF[l - 1][ti][kc], (short)0, acc,
                            false, false);
                    }
                }
                // scatter C fragment to LDS gates[col][row] (contiguous rows)
                int col = (wave * 2 + ti) * 16 + (lane & 15);
                int rb  = (lane >> 4) * 8;
                float* g = &gates[col * TILE_B + rb];
#pragma unroll
                for (int i = 0; i < 8; ++i) g[i] = acc[i];
            }
            __syncthreads();

            // ---- elementwise gate combine: c,h update ----
            {
                float x0 = xst[crow * 2 + 0], x1 = xst[crow * 2 + 1];
                _Float16 hnew[4];
#pragma unroll
                for (int i = 0; i < 4; ++i) {
                    int j = cj + i;
                    float gi = gates[(0 * HDIM + j) * TILE_B + crow] + biasc[l][0 * HDIM + j];
                    float gf = gates[(1 * HDIM + j) * TILE_B + crow] + biasc[l][1 * HDIM + j];
                    float gg = gates[(2 * HDIM + j) * TILE_B + crow] + biasc[l][2 * HDIM + j];
                    float go = gates[(3 * HDIM + j) * TILE_B + crow] + biasc[l][3 * HDIM + j];
                    if (l == 0) {   // tiny IN=2 input GEMM folded in scalar
                        gi += x0 * wi0[(0 * HDIM + j) * 2] + x1 * wi0[(0 * HDIM + j) * 2 + 1];
                        gf += x0 * wi0[(1 * HDIM + j) * 2] + x1 * wi0[(1 * HDIM + j) * 2 + 1];
                        gg += x0 * wi0[(2 * HDIM + j) * 2] + x1 * wi0[(2 * HDIM + j) * 2 + 1];
                        go += x0 * wi0[(3 * HDIM + j) * 2] + x1 * wi0[(3 * HDIM + j) * 2 + 1];
                    }
                    float c = fast_sigmoid(gf) * cpr[l][i] +
                              fast_sigmoid(gi) * fast_tanh(gg);
                    cpr[l][i] = c;
                    hnew[i] = (_Float16)(fast_sigmoid(go) * fast_tanh(c));
                }
#pragma unroll
                for (int i = 0; i < 4; ++i)
                    hst[l][crow * HDIM + cj + i] = hnew[i];
            }
            __syncthreads();
        }
    }

    // ---- final projection: h2(T-1) @ Wout^T + bout -> [16, 11] ----
    if (tid < TILE_B * NOUT) {
        int row = tid / NOUT, o = tid % NOUT;
        float s = bout[o];
#pragma unroll 8
        for (int k = 0; k < HDIM; ++k)
            s += Wout[o * HDIM + k] * (float)hst[2][row * HDIM + k];
        out[(size_t)(b0 + row) * NOUT + o] = s;
    }
}

extern "C" void kernel_launch(void* const* d_in, const int* in_sizes, int n_in,
                              void* d_out, int out_size, void* d_ws, size_t ws_size,
                              hipStream_t stream) {
    (void)in_sizes; (void)n_in; (void)d_ws; (void)ws_size; (void)out_size;
    const float* x    = (const float*)d_in[0];
    const float* Wih0 = (const float*)d_in[1];
    const float* Whh0 = (const float*)d_in[2];
    const float* bih0 = (const float*)d_in[3];
    const float* bhh0 = (const float*)d_in[4];
    const float* Wih1 = (const float*)d_in[5];
    const float* Whh1 = (const float*)d_in[6];
    const float* bih1 = (const float*)d_in[7];
    const float* bhh1 = (const float*)d_in[8];
    const float* Wih2 = (const float*)d_in[9];
    const float* Whh2 = (const float*)d_in[10];
    const float* bih2 = (const float*)d_in[11];
    const float* bhh2 = (const float*)d_in[12];
    const float* Wout = (const float*)d_in[13];
    const float* bout = (const float*)d_in[14];
    float* out = (float*)d_out;

    dim3 grid(BATCH / TILE_B);   // 32 workgroups, batch-parallel
    dim3 block(256);             // 8 wave32 waves
    lstm3_wmma_persistent<<<grid, block, 0, stream>>>(
        x, Wih0, Whh0, bih0, bhh0, Wih1, Whh1, bih1, bhh1,
        Wih2, Whh2, bih2, bhh2, Wout, bout, out);
}
